// ARSen_4707284156675
// MI455X (gfx1250) — matrix operand order repaired and checked by reading, not compiled, not run on hardware
//
#include <hip/hip_runtime.h>
#include <hip/hip_bf16.h>

// Additive attention, fused for MI455X (gfx1250, wave32, WMMA bf16).
//
// Math:  h = relu(keys @ W1_bot + (q @ W1_top + b1));  scores = h @ W2 (+b2, drops out
// of softmax);  alphas = softmax_S(scores);  context = alphas^T @ values.
//
// Dominant GEMM: 32768 x 1024 x 1024 (68.7 GFLOP) via v_wmma_f32_16x16x32_bf16 with
// fp32 accumulate.  keys are converted fp32->bf16 in registers using a single
// v_perm_b32 per packed pair (+0x8000 round bias), W1_bot is prepacked to bf16
// fragments once per launch (2 MB, L2-resident).  ReLU + dot(W2) fused into the GEMM
// epilogue so h (128 MB) never touches memory.

#define BATCH 16
#define SEQ   2048
#define HID   1024

typedef __attribute__((ext_vector_type(16))) __bf16 v16bf;
typedef __attribute__((ext_vector_type(8)))  float  v8f;

union Frag16 {                 // 32 bytes: one WMMA 16-bit A/B operand per lane
    unsigned int u[8];
    v16bf        v;
};

// Pack two fp32 into {bf16(b):bf16(a)} with one v_perm_b32.
// V_PERM_B32 byte source: S1 = bytes 0..3, S0 = bytes 4..7; sel 0x07060302 picks
// S1[3:2] into result[1:0] and S0[3:2] into result[3:2].
__device__ __forceinline__ unsigned int pack_bf16(float a, float b) {
    const unsigned int ua = __float_as_uint(a) + 0x8000u;   // round-to-nearest (ties up)
    const unsigned int ub = __float_as_uint(b) + 0x8000u;
    return __builtin_amdgcn_perm(ub, ua, 0x07060302u);
}

__device__ __forceinline__ v8f wmma_bf16(v16bf a, v16bf b, v8f c) {
    return __builtin_amdgcn_wmma_f32_16x16x32_bf16(
        /*neg_a=*/false, a, /*neg_b=*/false, b,
        /*c_mod=*/(short)0, c, /*reuse_a=*/false, /*reuse_b=*/false);
}

// ---------------------------------------------------------------------------
// Stage 1: pack W1 bottom half (rows H..2H-1) into bf16 B-matrix fragments.
// Fragment (tk, tn) covers K=[tk*32,tk*32+32) x N=[tn*16,tn*16+16).
// Lane l holds column n = tn*16 + (l&15), K values tk*32 + (l>>4)*8 + (i&7) + (i>>3)*16, i=0..15,
// packed pairwise into 8 dwords (32 B per lane, lanes contiguous).
// ---------------------------------------------------------------------------
__global__ void prep_w1_kernel(const float* __restrict__ W1, uint4* __restrict__ w1b) {
    const int t    = blockIdx.x * blockDim.x + threadIdx.x;   // 65536 threads
    const int lane = t & 31;
    const int tk   = (t >> 5) & 31;                           // 32 K-tiles
    const int tn   = t >> 10;                                 // 64 N-tiles
    const int n    = tn * 16 + (lane & 15);
    const int kb   = tk * 32 + (lane >> 4) * 8;

    unsigned int u[8];
#pragma unroll
    for (int j = 0; j < 8; ++j) {
        const float a = W1[(size_t)(HID + kb + 2 * j + (j >> 2) * 8)     * HID + n];
        const float b = W1[(size_t)(HID + kb + 2 * j + (j >> 2) * 8 + 1) * HID + n];
        u[j] = pack_bf16(a, b);
    }
    const size_t base = ((size_t)(tn * 32 + tk) * 32 + lane) * 2;   // in uint4
    w1b[base + 0] = make_uint4(u[0], u[1], u[2], u[3]);
    w1b[base + 1] = make_uint4(u[4], u[5], u[6], u[7]);
}

// ---------------------------------------------------------------------------
// Stage 2: qb[b][n] = b1[n] + sum_k q[b][k] * W1[k][n]   (top half of W1)
// ---------------------------------------------------------------------------
__global__ void prep_qb_kernel(const float* __restrict__ q,
                               const float* __restrict__ W1,
                               const float* __restrict__ b1,
                               float* __restrict__ qb) {
    const int t = blockIdx.x * blockDim.x + threadIdx.x;      // 16384 threads
    const int b = t >> 10;
    const int n = t & (HID - 1);
    float acc = b1[n];
    const float* qr = q + (size_t)b * HID;
#pragma unroll 4
    for (int k = 0; k < HID; ++k)
        acc = fmaf(qr[k], W1[(size_t)k * HID + n], acc);
    qb[(size_t)b * HID + n] = acc;
}

// ---------------------------------------------------------------------------
// Stage 3: GEMM keys @ W1_bot (bf16 WMMA, fp32 acc) with fused relu()*W2 epilogue.
// Grid: (S/128, B, H/512).  Block: 512 threads = 16 waves as 4(M) x 4(N).
// Each wave owns a 32(M) x 128(N) slab: 2x8 = 16 f32 16x16 accumulators.
// Partial row scores accumulated via global_atomic_add_f32.
// ---------------------------------------------------------------------------
__global__ __launch_bounds__(512)
void score_gemm_kernel(const float* __restrict__ keys,
                       const uint4* __restrict__ w1b,
                       const float* __restrict__ qb,
                       const float* __restrict__ W2,
                       float* __restrict__ scores) {
    const int tid  = threadIdx.x;
    const int lane = tid & 31;
    const int wave = tid >> 5;
    const int wm   = wave & 3;          // M position of wave in block
    const int wn   = wave >> 2;         // N position of wave in block
    const int b    = blockIdx.y;

    const int m_base = blockIdx.x * 128 + wm * 32;
    const int n_base = blockIdx.z * 512 + wn * 128;

    const int half = lane >> 4;         // 0: K 0..7/16..23 ; 1: K 8..15/24..31
    const int lidx = lane & 15;

    v8f acc[2][8];
#pragma unroll
    for (int mt = 0; mt < 2; ++mt)
#pragma unroll
        for (int nt = 0; nt < 8; ++nt)
#pragma unroll
            for (int e = 0; e < 8; ++e) acc[mt][nt][e] = 0.0f;

    // A-matrix row pointers (16-bit A layout: lane holds row m, K in 8+8 pattern)
    const float* arow0 = keys + ((size_t)b * SEQ + (m_base + lidx))      * HID + half * 8;
    const float* arow1 = keys + ((size_t)b * SEQ + (m_base + 16 + lidx)) * HID + half * 8;
    const uint4* bbase = w1b + (size_t)lane * 2;

    for (int k0 = 0; k0 < HID; k0 += 32) {
        __builtin_prefetch(arow0 + k0 + 64, 0, 0);   // global_prefetch_b8 K-slab two ahead
        Frag16 a0, a1;
        {
            const float4* p = (const float4*)(arow0 + k0);
            float4 x0 = p[0], x1 = p[1], x2 = p[4], x3 = p[5];
            a0.u[0] = pack_bf16(x0.x, x0.y); a0.u[1] = pack_bf16(x0.z, x0.w);
            a0.u[2] = pack_bf16(x1.x, x1.y); a0.u[3] = pack_bf16(x1.z, x1.w);
            a0.u[4] = pack_bf16(x2.x, x2.y); a0.u[5] = pack_bf16(x2.z, x2.w);
            a0.u[6] = pack_bf16(x3.x, x3.y); a0.u[7] = pack_bf16(x3.z, x3.w);
        }
        {
            const float4* p = (const float4*)(arow1 + k0);
            float4 x0 = p[0], x1 = p[1], x2 = p[4], x3 = p[5];
            a1.u[0] = pack_bf16(x0.x, x0.y); a1.u[1] = pack_bf16(x0.z, x0.w);
            a1.u[2] = pack_bf16(x1.x, x1.y); a1.u[3] = pack_bf16(x1.z, x1.w);
            a1.u[4] = pack_bf16(x2.x, x2.y); a1.u[5] = pack_bf16(x2.z, x2.w);
            a1.u[6] = pack_bf16(x3.x, x3.y); a1.u[7] = pack_bf16(x3.z, x3.w);
        }
        const int tk = k0 >> 5;
#pragma unroll
        for (int nt = 0; nt < 8; ++nt) {
            const int tn = (n_base >> 4) + nt;
            const uint4* bp = bbase + (size_t)(tn * 32 + tk) * 64;
            Frag16 bf;
            uint4 lo = bp[0], hi = bp[1];
            bf.u[0] = lo.x; bf.u[1] = lo.y; bf.u[2] = lo.z; bf.u[3] = lo.w;
            bf.u[4] = hi.x; bf.u[5] = hi.y; bf.u[6] = hi.z; bf.u[7] = hi.w;
            acc[0][nt] = wmma_bf16(a0.v, bf.v, acc[0][nt]);
            acc[1][nt] = wmma_bf16(a1.v, bf.v, acc[1][nt]);
        }
    }

    // Fused epilogue: part[row] = sum_n relu(acc + qb[b][n]) * W2[n]
    // C layout: lane l, vgpr e -> row = e + (l>>4)*8, col = (l&15).
    const float* qbrow = qb + (size_t)b * HID;
    float w2v[8], qv[8];
#pragma unroll
    for (int nt = 0; nt < 8; ++nt) {
        const int n = n_base + nt * 16 + lidx;
        w2v[nt] = W2[n];
        qv[nt]  = qbrow[n];
    }
    float part[2][8];
#pragma unroll
    for (int mt = 0; mt < 2; ++mt)
#pragma unroll
        for (int e = 0; e < 8; ++e) part[mt][e] = 0.0f;
#pragma unroll
    for (int mt = 0; mt < 2; ++mt)
#pragma unroll
        for (int nt = 0; nt < 8; ++nt)
#pragma unroll
            for (int e = 0; e < 8; ++e)
                part[mt][e] = fmaf(fmaxf(acc[mt][nt][e] + qv[nt], 0.0f), w2v[nt], part[mt][e]);

    // Reduce over the 16 lanes sharing a row (lanes 0-15: rows 0-7, lanes 16-31: rows 8-15)
#pragma unroll
    for (int mt = 0; mt < 2; ++mt)
#pragma unroll
        for (int e = 0; e < 8; ++e) {
            float s = part[mt][e];
            s += __shfl_xor(s, 1, 32);
            s += __shfl_xor(s, 2, 32);
            s += __shfl_xor(s, 4, 32);
            s += __shfl_xor(s, 8, 32);
            part[mt][e] = s;
        }
    if (lidx == 0) {
#pragma unroll
        for (int mt = 0; mt < 2; ++mt)
#pragma unroll
            for (int e = 0; e < 8; ++e)
                atomicAdd(&scores[(size_t)b * SEQ + m_base + mt * 16 + half * 8 + e],
                          part[mt][e]);
    }
}

// ---------------------------------------------------------------------------
// Stage 4: per-batch softmax over S=2048 (b2 is a uniform shift -> drops out).
// ---------------------------------------------------------------------------
__global__ void softmax_kernel(const float* __restrict__ scores,
                               float* __restrict__ alphas) {
    const int b   = blockIdx.x;
    const int tid = threadIdx.x;            // 256 threads, 8 values each
    float v[8];
#pragma unroll
    for (int i = 0; i < 8; ++i) v[i] = scores[(size_t)b * SEQ + i * 256 + tid];

    float m = v[0];
#pragma unroll
    for (int i = 1; i < 8; ++i) m = fmaxf(m, v[i]);
#pragma unroll
    for (int off = 16; off > 0; off >>= 1) m = fmaxf(m, __shfl_xor(m, off, 32));
    __shared__ float redm[8];
    if ((tid & 31) == 0) redm[tid >> 5] = m;
    __syncthreads();
    float bm = redm[0];
#pragma unroll
    for (int j = 1; j < 8; ++j) bm = fmaxf(bm, redm[j]);

    float s = 0.0f;
#pragma unroll
    for (int i = 0; i < 8; ++i) { v[i] = __expf(v[i] - bm); s += v[i]; }
#pragma unroll
    for (int off = 16; off > 0; off >>= 1) s += __shfl_xor(s, off, 32);
    __shared__ float reds[8];
    if ((tid & 31) == 0) reds[tid >> 5] = s;
    __syncthreads();
    float tot = 0.0f;
#pragma unroll
    for (int j = 0; j < 8; ++j) tot += reds[j];
    const float inv = 1.0f / tot;
#pragma unroll
    for (int i = 0; i < 8; ++i)
        alphas[(size_t)b * SEQ + i * 256 + tid] = v[i] * inv;
}

// ---------------------------------------------------------------------------
// Stage 5: context[b][:] = sum_s alphas[b][s] * values[b][s][:]   (128 MB pass)
// Grid: (B, S/256). Alphas chunk staged in LDS, float4 accumulation, atomic merge.
// ---------------------------------------------------------------------------
__global__ void context_kernel(const float* __restrict__ values,
                               const float* __restrict__ alphas,
                               float* __restrict__ ctx) {
    const int b   = blockIdx.x;
    const int sc  = blockIdx.y;             // seq chunk of 256
    const int tid = threadIdx.x;            // 256 threads
    __shared__ float sa[256];
    sa[tid] = alphas[(size_t)b * SEQ + sc * 256 + tid];
    __syncthreads();

    const int h0 = tid * 4;
    float4 accv = make_float4(0.f, 0.f, 0.f, 0.f);
    const float4* vp =
        (const float4*)(values + ((size_t)b * SEQ + (size_t)sc * 256) * HID + h0);
#pragma unroll 4
    for (int j = 0; j < 256; ++j) {
        const float4 x = vp[(size_t)j * (HID / 4)];
        const float  a = sa[j];
        accv.x = fmaf(a, x.x, accv.x);
        accv.y = fmaf(a, x.y, accv.y);
        accv.z = fmaf(a, x.z, accv.z);
        accv.w = fmaf(a, x.w, accv.w);
    }
    atomicAdd(&ctx[(size_t)b * HID + h0 + 0], accv.x);
    atomicAdd(&ctx[(size_t)b * HID + h0 + 1], accv.y);
    atomicAdd(&ctx[(size_t)b * HID + h0 + 2], accv.z);
    atomicAdd(&ctx[(size_t)b * HID + h0 + 3], accv.w);
}

// ---------------------------------------------------------------------------
extern "C" void kernel_launch(void* const* d_in, const int* in_sizes, int n_in,
                              void* d_out, int out_size, void* d_ws, size_t ws_size,
                              hipStream_t stream) {
    const float* q    = (const float*)d_in[0];   // [16,1024]
    const float* keys = (const float*)d_in[1];   // [16,2048,1024]
    const float* vals = (const float*)d_in[2];   // [16,2048,1024]
    const float* W1   = (const float*)d_in[3];   // [2048,1024]
    const float* b1   = (const float*)d_in[4];   // [1024]
    const float* W2   = (const float*)d_in[5];   // [1024]
    // d_in[6] = b2 : uniform shift, invariant under softmax -> unused.

    // Workspace layout (~2.2 MB): bf16 W1 fragments | qb | scores
    char*  ws     = (char*)d_ws;
    uint4* w1b    = (uint4*)ws;                                    // 2 MB
    float* qb     = (float*)(ws + (2u << 20));                     // 64 KB
    float* scores = (float*)(ws + (2u << 20) + (64u << 10));       // 128 KB

    float* ctx    = (float*)d_out;                                 // [16,1,1024]
    float* alphas = (float*)d_out + BATCH * HID;                   // [16,2048,1]

    hipMemsetAsync(scores, 0, (size_t)BATCH * SEQ * sizeof(float), stream);
    hipMemsetAsync(ctx,    0, (size_t)BATCH * HID * sizeof(float), stream);

    prep_w1_kernel<<<256, 256, 0, stream>>>(W1, w1b);
    prep_qb_kernel<<<64, 256, 0, stream>>>(q, W1, b1, qb);
    score_gemm_kernel<<<dim3(SEQ / 128, BATCH, HID / 512), 512, 0, stream>>>(
        keys, w1b, qb, W2, scores);
    softmax_kernel<<<BATCH, 256, 0, stream>>>(scores, alphas);
    context_kernel<<<dim3(BATCH, SEQ / 256), 256, 0, stream>>>(vals, alphas, ctx);
}
